// EpipolarCrossAttention_51376398794963
// MI455X (gfx1250) — matrix operand order repaired and checked by reading, not compiled, not run on hardware
//
#include <hip/hip_runtime.h>

typedef __bf16 bf16;
typedef __attribute__((ext_vector_type(16))) __bf16 v16bf;
typedef __attribute__((ext_vector_type(8)))  float  v8f;
typedef __attribute__((ext_vector_type(8)))  unsigned int v8u;

constexpr int C_DIM = 256;
constexpr int LDQ   = C_DIM + 8;   // padded LDS row stride (bf16 elems)

#define WMMA_BF16(A, B, Cacc) \
  __builtin_amdgcn_wmma_f32_16x16x32_bf16(false, (A), false, (B), (short)0, (Cacc), false, false)

#if __has_builtin(__builtin_amdgcn_global_load_async_to_lds_b128)
#define HAVE_ASYNC_LDS 1
#endif

// clang builtin signature: void(v4i addrspace(1)*, v4i addrspace(3)*, imm int, imm int)
typedef int v4i_b128 __attribute__((vector_size(16)));
typedef __attribute__((address_space(1))) v4i_b128* g_b128_ptr;
typedef __attribute__((address_space(3))) v4i_b128* l_b128_ptr;

// memory -> LDS, 16 bytes per lane. Async (no VGPR round trip, ASYNCcnt) when available.
__device__ __forceinline__ void copy16_to_lds(const bf16* g, bf16* l) {
#if defined(HAVE_ASYNC_LDS)
  __builtin_amdgcn_global_load_async_to_lds_b128(
      (g_b128_ptr)(v4i_b128*)g, (l_b128_ptr)(v4i_b128*)l, 0, 0);
#else
  *(uint4*)l = *(const uint4*)g;
#endif
}

__device__ __forceinline__ void wait_async_copies() {
#if defined(HAVE_ASYNC_LDS)
#if __has_builtin(__builtin_amdgcn_s_wait_asynccnt)
  __builtin_amdgcn_s_wait_asynccnt(0);
#else
  asm volatile("s_wait_asynccnt 0x0" ::: "memory");
#endif
#endif
}

// ---- WMMA fragment loaders (contiguous k order: element s of lane half h is k = 16h + s, wave32) ----

// A matrix 16x32 bf16, src row-major [m][k] with leading dim ld (pairs -> b32 loads).
__device__ __forceinline__ v16bf load_a_frag(const bf16* base, int ld) {
  const int l  = threadIdx.x & 31;
  const int m  = l & 15;
  const int hi = l >> 4;
  const bf16* row = base + m * ld;
  v8u a;
#pragma unroll
  for (int r = 0; r < 8; ++r) {
    const int kk = (hi << 4) + (r << 1);
    a[r] = *(const unsigned int*)(row + kk);
  }
  return __builtin_bit_cast(v16bf, a);
}

// B matrix 32x16 where B[k][n] = src[n][k] (transposed operand; pairs -> b32 loads).
__device__ __forceinline__ v16bf load_bt_frag(const bf16* base, int ld) {
  const int l  = threadIdx.x & 31;
  const int n  = l & 15;
  const int hi = l >> 4;
  const bf16* row = base + n * ld;
  v8u b;
#pragma unroll
  for (int r = 0; r < 8; ++r) {
    const int kk = (hi << 4) + (r << 1);
    b[r] = *(const unsigned int*)(row + kk);
  }
  return __builtin_bit_cast(v16bf, b);
}

// Same as load_bt_frag but source is f32 (convert to bf16 on the fly).
__device__ __forceinline__ v16bf load_bt_frag_f32(const float* base, int ld) {
  const int l  = threadIdx.x & 31;
  const int n  = l & 15;
  const int hi = l >> 4;
  v16bf b;
#pragma unroll
  for (int r = 0; r < 8; ++r) {
    const int kk = (hi << 4) + (r << 1);
    b[2*r]   = (bf16)base[n*ld + kk];
    b[2*r+1] = (bf16)base[n*ld + kk + 1];
  }
  return b;
}

// B matrix 32x16 where B[k][n] = src[k][n] (row-major K x N; column reads).
__device__ __forceinline__ v16bf load_b_frag(const bf16* base, int ld) {
  const int l  = threadIdx.x & 31;
  const int n  = l & 15;
  const int hi = l >> 4;
  v16bf b;
#pragma unroll
  for (int r = 0; r < 8; ++r) {
    const int kk = (hi << 4) + (r << 1);
    b[2*r]   = base[kk*ld + n];
    b[2*r+1] = base[(kk + 1)*ld + n];
  }
  return b;
}

// ---- Kernel 1: Y(bf16)[rows,C] = X(f32)[rows,C] @ W(f32)[C,C]^T + bias ----
__global__ __launch_bounds__(256) void proj_f32_to_bf16(
    const float* __restrict__ X, const float* __restrict__ W,
    const float* __restrict__ bias, bf16* __restrict__ Y)
{
  __shared__ __attribute__((aligned(16))) bf16 Xs[64][LDQ];
  const int tid = threadIdx.x;
  const size_t rowBase = (size_t)blockIdx.x * 64;

  for (int i = tid; i < 64 * (C_DIM / 4); i += 256) {
    const int r  = i >> 6;
    const int c4 = (i & 63) << 2;
    const float4 v = *(const float4*)(X + (rowBase + r) * C_DIM + c4);
    Xs[r][c4+0] = (bf16)v.x; Xs[r][c4+1] = (bf16)v.y;
    Xs[r][c4+2] = (bf16)v.z; Xs[r][c4+3] = (bf16)v.w;
  }
  __syncthreads();

  const int wave   = tid >> 5;
  const int rt     = wave & 3;          // row tile (16 rows)
  const int ctBase = (wave >> 2) * 8;   // 8 col tiles of 16
  const v8f vzero = {0.f,0.f,0.f,0.f,0.f,0.f,0.f,0.f};
  v8f acc[8];
#pragma unroll
  for (int t = 0; t < 8; ++t) acc[t] = vzero;

  for (int ks = 0; ks < C_DIM; ks += 32) {
    const v16bf a = load_a_frag(&Xs[rt * 16][ks], LDQ);
#pragma unroll
    for (int t = 0; t < 8; ++t) {
      const int c0 = (ctBase + t) * 16;
      const v16bf b = load_bt_frag_f32(W + (size_t)c0 * C_DIM + ks, C_DIM);
      acc[t] = WMMA_BF16(a, b, acc[t]);
    }
  }

  const int lane = tid & 31;
  const int nI   = lane & 15;
  const int hiL  = lane >> 4;
#pragma unroll
  for (int t = 0; t < 8; ++t) {
    const int c = (ctBase + t) * 16 + nI;
    const float bb = bias[c];
#pragma unroll
    for (int r = 0; r < 8; ++r) {
      const size_t row = rowBase + rt * 16 + hiL * 8 + r;
      Y[row * C_DIM + c] = (bf16)(acc[t][r] + bb);
    }
  }
}

// ---- Kernel 3: Y(f32)[rows,C] = X(bf16)[rows,C] @ W(f32)[C,C]^T + bias ----
__global__ __launch_bounds__(256) void proj_bf16_to_f32(
    const bf16* __restrict__ X, const float* __restrict__ W,
    const float* __restrict__ bias, float* __restrict__ Y)
{
  __shared__ __attribute__((aligned(16))) bf16 Xs[64][LDQ];
  const int tid = threadIdx.x;
  const size_t rowBase = (size_t)blockIdx.x * 64;

  for (int i = tid; i < 64 * (C_DIM / 8); i += 256) {
    const int r  = i >> 5;
    const int c8 = (i & 31) << 3;
    copy16_to_lds(X + (rowBase + r) * C_DIM + c8, &Xs[r][c8]);
  }
  wait_async_copies();
  __syncthreads();

  const int wave   = tid >> 5;
  const int rt     = wave & 3;
  const int ctBase = (wave >> 2) * 8;
  const v8f vzero = {0.f,0.f,0.f,0.f,0.f,0.f,0.f,0.f};
  v8f acc[8];
#pragma unroll
  for (int t = 0; t < 8; ++t) acc[t] = vzero;

  for (int ks = 0; ks < C_DIM; ks += 32) {
    const v16bf a = load_a_frag(&Xs[rt * 16][ks], LDQ);
#pragma unroll
    for (int t = 0; t < 8; ++t) {
      const int c0 = (ctBase + t) * 16;
      const v16bf b = load_bt_frag_f32(W + (size_t)c0 * C_DIM + ks, C_DIM);
      acc[t] = WMMA_BF16(a, b, acc[t]);
    }
  }

  const int lane = tid & 31;
  const int nI   = lane & 15;
  const int hiL  = lane >> 4;
#pragma unroll
  for (int t = 0; t < 8; ++t) {
    const int c = (ctBase + t) * 16 + nI;
    const float bb = bias[c];
#pragma unroll
    for (int r = 0; r < 8; ++r) {
      const size_t row = rowBase + rt * 16 + hiL * 8 + r;
      Y[row * C_DIM + c] = acc[t][r] + bb;
    }
  }
}

// ---- Kernel 2: fused epipolar flash-attention, double-buffered async K/V ----
__global__ __launch_bounds__(256) void epi_attn(
    const bf16* __restrict__ Qg, const bf16* __restrict__ Kg, const bf16* __restrict__ Vg,
    const float* __restrict__ kptsL, const float* __restrict__ kptsR,
    bf16* __restrict__ Mout, float* __restrict__ dispOut, float* __restrict__ confOut,
    int Nn, int Mm)
{
  __shared__ __attribute__((aligned(16))) bf16 Qs[64][LDQ];
  __shared__ __attribute__((aligned(16))) bf16 Ks[2][64][LDQ];
  __shared__ __attribute__((aligned(16))) bf16 Vs[2][64][LDQ];
  __shared__ __attribute__((aligned(16))) bf16 Ps[64][72];
  __shared__ float Ss[64][68];
  __shared__ float qu[64], qv[64], ku[2][64], kv[2][64];
  __shared__ float m_s[64], l_s[64], d_s[64], cf_s[64];
  __shared__ float red_mx[64][4], red_sum[64][4], red_dsum[64][4];
  __shared__ int   red_any[64][4];
  __shared__ int   any_s[64];

  const int tid = threadIdx.x;
  const int b = blockIdx.y;
  const size_t qBase = (size_t)blockIdx.x * 64;
  const size_t qRow0 = (size_t)b * Nn + qBase;

  const int wave   = tid >> 5;
  const int lane   = tid & 31;
  const int rt     = wave & 3;          // query row tile
  const int ctBase = (wave >> 2) * 8;   // output-channel tiles (P@V)
  const int ctp    = (wave >> 2) * 2;   // key-col tile pair (Q@K^T)
  const int nI     = lane & 15;
  const int hiL    = lane >> 4;
  const float scale = 0.0625f;          // 1/sqrt(256)

  // stage next key/value chunk into buffer p (async when available)
  auto stage_chunk = [&](int p, int jb) {
    const size_t kRow0 = (size_t)b * Mm + jb;
    for (int i = tid; i < 64 * (C_DIM / 8); i += 256) {
      const int r  = i >> 5;
      const int c8 = (i & 31) << 3;
      copy16_to_lds(Kg + (kRow0 + r) * C_DIM + c8, &Ks[p][r][c8]);
      copy16_to_lds(Vg + (kRow0 + r) * C_DIM + c8, &Vs[p][r][c8]);
    }
    if (tid < 64) {
      ku[p][tid] = kptsR[(kRow0 + tid) * 2 + 0];
      kv[p][tid] = kptsR[(kRow0 + tid) * 2 + 1];
    }
  };

  // stage Q + init per-row state + preload chunk 0
  for (int i = tid; i < 64 * (C_DIM / 8); i += 256) {
    const int r  = i >> 5;
    const int c8 = (i & 31) << 3;
    copy16_to_lds(Qg + (qRow0 + r) * C_DIM + c8, &Qs[r][c8]);
  }
  if (tid < 64) {
    qu[tid] = kptsL[(qRow0 + tid) * 2 + 0];
    qv[tid] = kptsL[(qRow0 + tid) * 2 + 1];
    m_s[tid] = -3.0e38f;
    l_s[tid] = 0.f;
    d_s[tid] = 0.f;
    any_s[tid] = 0;
  }
  stage_chunk(0, 0);
  wait_async_copies();
  __syncthreads();

  const v8f vzero = {0.f,0.f,0.f,0.f,0.f,0.f,0.f,0.f};
  v8f acc[8];
#pragma unroll
  for (int t = 0; t < 8; ++t) acc[t] = vzero;

  const int srow = tid >> 2;            // softmax row 0..63
  const int seg  = tid & 3;             // 16-key segment 0..3

  int p = 0;
  for (int jb = 0; jb < Mm; jb += 64, p ^= 1) {
    // prefetch next chunk into the other buffer (its last reads precede the loop-end barrier)
    if (jb + 64 < Mm) stage_chunk(p ^ 1, jb + 64);

    // S = scale * Q @ K^T  (64x64, 16 tiles over 8 waves)
#pragma unroll
    for (int t = 0; t < 2; ++t) {
      const int ct = ctp + t;
      v8f s = vzero;
      for (int ks = 0; ks < C_DIM; ks += 32) {
        const v16bf a  = load_a_frag(&Qs[rt * 16][ks], LDQ);
        const v16bf bb = load_bt_frag(&Ks[p][ct * 16][ks], LDQ);
        s = WMMA_BF16(a, bb, s);
      }
#pragma unroll
      for (int r = 0; r < 8; ++r)
        Ss[rt * 16 + hiL * 8 + r][ct * 16 + nI] = s[r] * scale;
    }
    __syncthreads();

    // masked online softmax (4 threads per row), -1e9 sentinel as in reference
    float sv[16];
    const float uL = qu[srow];
    const float vL = qv[srow];
    float mloc = -3.0e38f;
    int   aloc = 0;
#pragma unroll
    for (int j = 0; j < 16; ++j) {
      const int jj = seg * 16 + j;
      const float du = uL - ku[p][jj];
      const float dv = fabsf(vL - kv[p][jj]);
      const bool ok = (dv < 3.0f) && (du > 0.0f) && (du < 192.0f);
      const float s = ok ? Ss[srow][jj] : -1.0e9f;
      sv[j] = s;
      mloc = fmaxf(mloc, s);
      aloc |= ok ? 1 : 0;
    }
    red_mx[srow][seg]  = mloc;
    red_any[srow][seg] = aloc;
    __syncthreads();
    const float mold = m_s[srow];
    float mnew = fmaxf(fmaxf(red_mx[srow][0], red_mx[srow][1]),
                       fmaxf(red_mx[srow][2], red_mx[srow][3]));
    mnew = fmaxf(mold, mnew);
    const float cf = __expf(mold - mnew);
    float psum = 0.f, dsum = 0.f;
#pragma unroll
    for (int j = 0; j < 16; ++j) {
      const int jj = seg * 16 + j;
      const float pj = __expf(sv[j] - mnew);
      Ps[srow][jj] = (bf16)pj;
      psum += pj;
      dsum += pj * (uL - ku[p][jj]);
    }
    red_sum[srow][seg]  = psum;
    red_dsum[srow][seg] = dsum;
    __syncthreads();
    if (seg == 0) {
      cf_s[srow] = cf;
      m_s[srow] = mnew;
      l_s[srow] = l_s[srow] * cf +
                  red_sum[srow][0] + red_sum[srow][1] + red_sum[srow][2] + red_sum[srow][3];
      d_s[srow] = d_s[srow] * cf +
                  red_dsum[srow][0] + red_dsum[srow][1] + red_dsum[srow][2] + red_dsum[srow][3];
      any_s[srow] |= red_any[srow][0] | red_any[srow][1] | red_any[srow][2] | red_any[srow][3];
    }
    __syncthreads();

    // rescale O accumulators, then O += P @ V
    float cfr[8];
#pragma unroll
    for (int r = 0; r < 8; ++r) cfr[r] = cf_s[rt * 16 + hiL * 8 + r];
#pragma unroll
    for (int t = 0; t < 8; ++t)
#pragma unroll
      for (int r = 0; r < 8; ++r) acc[t][r] *= cfr[r];

#pragma unroll
    for (int kk = 0; kk < 64; kk += 32) {
      const v16bf a = load_a_frag(&Ps[rt * 16][kk], 72);
#pragma unroll
      for (int t = 0; t < 8; ++t) {
        const v16bf bb = load_b_frag(&Vs[p][kk][(ctBase + t) * 16], LDQ);
        acc[t] = WMMA_BF16(a, bb, acc[t]);
      }
    }

    // next chunk's async copies must have landed; all waves done with buffer p
    wait_async_copies();
    __syncthreads();
  }

  if (tid < 64) {
    const float inv = 1.0f / l_s[tid];
    dispOut[qRow0 + tid] = d_s[tid] * inv;
    confOut[qRow0 + tid] = any_s[tid] ? 1.0f : 0.0f;
    cf_s[tid] = inv;                    // reuse as normalization factor
  }
  __syncthreads();
#pragma unroll
  for (int t = 0; t < 8; ++t) {
    const int c = (ctBase + t) * 16 + nI;
#pragma unroll
    for (int r = 0; r < 8; ++r) {
      const int row = rt * 16 + hiL * 8 + r;
      Mout[(qRow0 + row) * C_DIM + c] = (bf16)(acc[t][r] * cf_s[row]);
    }
  }
}

extern "C" void kernel_launch(void* const* d_in, const int* in_sizes, int n_in,
                              void* d_out, int out_size, void* d_ws, size_t ws_size,
                              hipStream_t stream) {
  (void)in_sizes; (void)n_in; (void)out_size; (void)ws_size;
  const float* nodes_L = (const float*)d_in[0];
  const float* nodes_R = (const float*)d_in[1];
  const float* kpts_L  = (const float*)d_in[2];
  const float* kpts_R  = (const float*)d_in[3];
  const float* Wq = (const float*)d_in[4];
  const float* bq = (const float*)d_in[5];
  const float* Wk = (const float*)d_in[6];
  const float* bk = (const float*)d_in[7];
  const float* Wv = (const float*)d_in[8];
  const float* bv = (const float*)d_in[9];
  const float* Wm = (const float*)d_in[10];
  const float* bm = (const float*)d_in[11];

  const int B = 4, N = 4096, M = 4096;
  const size_t tokN = (size_t)B * N;
  const size_t tokM = (size_t)B * M;

  // workspace: Q, K, V, matched as bf16 (4 x 8 MB = 32 MB)
  bf16* Qw = (bf16*)d_ws;
  bf16* Kw = Qw + tokN * C_DIM;
  bf16* Vw = Kw + tokM * C_DIM;
  bf16* Mw = Vw + tokM * C_DIM;

  float* out = (float*)d_out;
  float* dispOut = out + tokN * C_DIM;
  float* confOut = dispOut + tokN;

  proj_f32_to_bf16<<<(int)(tokN / 64), 256, 0, stream>>>(nodes_L, Wq, bq, Qw);
  proj_f32_to_bf16<<<(int)(tokM / 64), 256, 0, stream>>>(nodes_R, Wk, bk, Kw);
  proj_f32_to_bf16<<<(int)(tokM / 64), 256, 0, stream>>>(nodes_R, Wv, bv, Vw);

  epi_attn<<<dim3(N / 64, B), 256, 0, stream>>>(Qw, Kw, Vw, kpts_L, kpts_R,
                                                Mw, dispOut, confOut, N, M);

  proj_bf16_to_f32<<<(int)(tokN / 64), 256, 0, stream>>>(Mw, Wm, bm, out);
}
